// Net_45337674776577
// MI455X (gfx1250) — compile-verified, matrix operand order checked by reading, not run on hardware
//
#include <hip/hip_runtime.h>
#include <hip/hip_bf16.h>

#define NN   50000           // nodes
#define EE   400000          // gat edges (before self loops)
#define ESZ  400000          // scatter nnz
#define NH   64              // hidden dim
#define HD   4               // heads
#define HF   (HD*NH)         // 256
#define NSCB 3
#define BG   64              // graphs
#define NC   10
#define EPSV 1e-5f
#define ET   (EE+NN)         // edges incl self loops

typedef __attribute__((ext_vector_type(16))) _Float16 v16h;
typedef __attribute__((ext_vector_type(8)))  float    v8f;

// ---------------------------------------------------------------- utilities
__global__ void k_fill(float* __restrict__ p, float v, size_t n) {
  size_t i = (size_t)blockIdx.x * blockDim.x + threadIdx.x;
  if (i < n) p[i] = v;
}

__device__ inline void atomicMaxFloat(float* addr, float val) {
  unsigned int* ua = (unsigned int*)addr;
  unsigned int old = *ua, assumed;
  do {
    if (__uint_as_float(old) >= val) return;
    assumed = old;
    old = atomicCAS(ua, assumed, __float_as_uint(val));
  } while (old != assumed);
}

// ------------------------------------------------- weight pre-pack (f32->f16)
// Pack W[Kin x Kout] into WMMA B-fragment order: for each (n-tile tn, k-step ks,
// lane), the 16 halves that lane feeds to v_wmma, contiguous (32B -> 2x b128).
__global__ void k_pack_w(const float* __restrict__ W, _Float16* __restrict__ Wp,
                         int Kin, int Kout) {
  int tid = blockIdx.x * blockDim.x + threadIdx.x;
  int ksteps = Kin >> 5;
  int total = (Kout >> 4) * ksteps * 32;
  if (tid >= total) return;
  int lane = tid & 31;
  int tmp  = tid >> 5;
  int ks = tmp % ksteps;
  int tn = tmp / ksteps;
  int hs = lane >> 4, r = lane & 15;
  int col = tn * 16 + r;
  int k0  = ks * 32 + hs * 8;                 // ISA 7.12.2 16-bit B layout
  _Float16* dst = Wp + (size_t)tid * 16;
  #pragma unroll
  for (int i = 0; i < 8; ++i) {
    dst[i]     = (_Float16)W[(size_t)(k0 + i) * Kout + col];
    dst[8 + i] = (_Float16)W[(size_t)(k0 + 16 + i) * Kout + col];
  }
}

// ------------------------------------------------------- WMMA tiled GEMM
// C[M x KOUT] = A[M x KIN] @ W + bias.  One wave computes a 16 x (16*TN) strip:
// A fragment loaded once per k-step (4x global_load_b128), fed to TN wmmas.
// B fragments come pre-packed (2x global_load_b128 each). M % 16 == 0.
template<int KIN, int KOUT, int TN>
__global__ __launch_bounds__(256)
void k_linear_t(const float* __restrict__ A, const _Float16* __restrict__ Wp,
                const float* __restrict__ bias, float* __restrict__ C, int M) {
  constexpr int KSTEPS = KIN / 32;
  constexpr int WPR    = KOUT / (16 * TN);    // waves per 16-row strip
  int wave   = blockIdx.x * (blockDim.x >> 5) + (threadIdx.x >> 5);
  int tilesM = M >> 4;
  if (wave >= tilesM * WPR) return;           // wave-uniform: EXEC stays all-1s
  int tm = wave / WPR;
  int tw = wave - tm * WPR;
  int lane = threadIdx.x & 31;
  int hs = lane >> 4, r = lane & 15;

  const float* arow = A + (size_t)(tm * 16 + r) * KIN + hs * 8;
  const v16h*  wp   = (const v16h*)(Wp + (size_t)(tw * TN) * KSTEPS * 32 * 16);

  v8f c[TN];
  #pragma unroll
  for (int t = 0; t < TN; ++t) c[t] = (v8f){};

  #pragma unroll
  for (int ks = 0; ks < KSTEPS; ++ks) {
    __builtin_prefetch(arow + ks * 32 + 64, 0, 0);   // global_prefetch_b8
    const float4* a0 = (const float4*)(arow + ks * 32);
    const float4* a1 = (const float4*)(arow + ks * 32 + 16);
    float4 p0 = a0[0], p1 = a0[1], q0 = a1[0], q1 = a1[1];
    v16h a;
    a[0]  = (_Float16)p0.x; a[1]  = (_Float16)p0.y;
    a[2]  = (_Float16)p0.z; a[3]  = (_Float16)p0.w;
    a[4]  = (_Float16)p1.x; a[5]  = (_Float16)p1.y;
    a[6]  = (_Float16)p1.z; a[7]  = (_Float16)p1.w;
    a[8]  = (_Float16)q0.x; a[9]  = (_Float16)q0.y;
    a[10] = (_Float16)q0.z; a[11] = (_Float16)q0.w;
    a[12] = (_Float16)q1.x; a[13] = (_Float16)q1.y;
    a[14] = (_Float16)q1.z; a[15] = (_Float16)q1.w;
    #pragma unroll
    for (int t = 0; t < TN; ++t) {
      v16h b = wp[(size_t)(t * KSTEPS + ks) * 32 + lane];
      c[t] = __builtin_amdgcn_wmma_f32_16x16x32_f16(false, a, false, b,
                                                    (short)0, c[t], false, false);
    }
  }
  // C/D layout: VGPR v -> row = tm*16 + hs*8 + v, col = tile*16 + r
  #pragma unroll
  for (int t = 0; t < TN; ++t) {
    int col  = (tw * TN + t) * 16 + r;
    float bv = bias ? bias[col] : 0.0f;
    int base = (tm * 16 + hs * 8) * KOUT + col;   // 32-bit index (fits < 2^31)
    #pragma unroll
    for (int v = 0; v < 8; ++v)
      C[base + v * KOUT] = c[t][v] + bv;
  }
}

// ---------------------------------------------------- sparse scatter matmul
__global__ void k_scatter(const int* __restrict__ ei, const float* __restrict__ attr,
                          const float* __restrict__ src_feat, float* __restrict__ out) {
  size_t gid = (size_t)blockIdx.x * blockDim.x + threadIdx.x;
  if (gid >= (size_t)ESZ * NH) return;
  int e = (int)(gid >> 6);
  int f = (int)(gid & 63);
  int s = ei[e];
  int d = ei[ESZ + e];
  atomicAdd(&out[d * NH + f], attr[e] * src_feat[s * NH + f]);
}

// ------------------------------------------------------------- GAT pieces
__global__ void k_att_scores(const float* __restrict__ hp,
                             const float* __restrict__ asrc,
                             const float* __restrict__ adst,
                             float* __restrict__ als, float* __restrict__ ald) {
  int gid = blockIdx.x * blockDim.x + threadIdx.x;
  if (gid >= NN * HD) return;
  int n = gid >> 2, h = gid & 3;
  const float* row = hp + (size_t)n * HF + h * NH;
  float s0 = 0.f, s1 = 0.f;
  for (int c = 0; c < NH; ++c) {
    float v = row[c];
    s0 += v * asrc[h * NH + c];
    s1 += v * adst[h * NH + c];
  }
  als[gid] = s0;
  ald[gid] = s1;
}

__device__ inline void edge_sd(const int* __restrict__ ei, int e, int& s, int& d) {
  if (e < EE) { s = ei[e]; d = ei[EE + e]; }
  else        { s = e - EE; d = e - EE; }      // appended self loops
}

__global__ void k_edge_pass1(const int* __restrict__ ei,
                             const float* __restrict__ als,
                             const float* __restrict__ ald,
                             float* __restrict__ ebuf, float* __restrict__ mmax) {
  size_t gid = (size_t)blockIdx.x * blockDim.x + threadIdx.x;
  if (gid >= (size_t)ET * HD) return;
  int e = (int)(gid >> 2), h = (int)(gid & 3);
  int s, d; edge_sd(ei, e, s, d);
  float v = als[s * HD + h] + ald[d * HD + h];
  v = v > 0.f ? v : 0.2f * v;                  // leaky_relu(0.2)
  ebuf[gid] = v;
  atomicMaxFloat(&mmax[d * HD + h], v);
}

__global__ void k_edge_pass2(const int* __restrict__ ei,
                             const float* __restrict__ mmax,
                             float* __restrict__ ebuf, float* __restrict__ denom) {
  size_t gid = (size_t)blockIdx.x * blockDim.x + threadIdx.x;
  if (gid >= (size_t)ET * HD) return;
  int e = (int)(gid >> 2), h = (int)(gid & 3);
  int s, d; edge_sd(ei, e, s, d);
  float v = __expf(ebuf[gid] - mmax[d * HD + h]);
  ebuf[gid] = v;
  atomicAdd(&denom[d * HD + h], v);
}

__global__ void k_edge_aggr(const int* __restrict__ ei,
                            const float* __restrict__ ebuf,
                            const float* __restrict__ denom,
                            const float* __restrict__ hp, float* __restrict__ gz) {
  size_t gid = (size_t)blockIdx.x * blockDim.x + threadIdx.x;
  if (gid >= (size_t)ET * HF) return;
  int e = (int)(gid >> 8);
  int c = (int)(gid & 255);
  int h = c >> 6;
  int s, d; edge_sd(ei, e, s, d);
  float alpha = ebuf[(size_t)e * HD + h] / denom[d * HD + h];
  atomicAdd(&gz[(size_t)d * HF + c], alpha * hp[(size_t)s * HF + c]);
}

__global__ void k_bias_elu(float* __restrict__ gz, const float* __restrict__ b) {
  size_t gid = (size_t)blockIdx.x * blockDim.x + threadIdx.x;
  if (gid >= (size_t)NN * HF) return;
  float v = gz[gid] + b[gid & 255];
  gz[gid] = v > 0.f ? v : (__expf(v) - 1.0f);  // ELU
}

// -------------------------------------------------------------- batch norm
__global__ void k_bn_stats(const float* __restrict__ z, float* __restrict__ stats,
                           int R, int C) {
  int c = blockIdx.x;                           // one block per channel
  __shared__ float s1[256], s2[256];
  float a = 0.f, b = 0.f;
  for (int r = threadIdx.x; r < R; r += blockDim.x) {
    float v = z[(size_t)r * C + c];
    a += v; b += v * v;
  }
  s1[threadIdx.x] = a; s2[threadIdx.x] = b;
  __syncthreads();
  for (int off = blockDim.x >> 1; off > 0; off >>= 1) {
    if ((int)threadIdx.x < off) {
      s1[threadIdx.x] += s1[threadIdx.x + off];
      s2[threadIdx.x] += s2[threadIdx.x + off];
    }
    __syncthreads();
  }
  if (threadIdx.x == 0) {
    float m = s1[0] / R;
    stats[c]     = m;
    stats[C + c] = s2[0] / R - m * m;           // biased variance
  }
}

__global__ void k_bn_apply_acc(const float* __restrict__ z,
                               const float* __restrict__ stats,
                               const float* __restrict__ gamma,
                               const float* __restrict__ beta,
                               float* __restrict__ acc) {
  size_t gid = (size_t)blockIdx.x * blockDim.x + threadIdx.x;
  if (gid >= (size_t)NN * NH) return;
  int c = (int)(gid & 63);
  float inv = rsqrtf(stats[NH + c] + EPSV);
  acc[gid] += (z[gid] - stats[c]) * inv * gamma[c] + beta[c];
}

__global__ void k_add(float* __restrict__ h, const float* __restrict__ acc, size_t n) {
  size_t gid = (size_t)blockIdx.x * blockDim.x + threadIdx.x;
  if (gid < n) h[gid] += acc[gid];
}

// --------------------------------------------------------------- head
__global__ void k_pool(const int* __restrict__ batch, const float* __restrict__ h,
                       float* __restrict__ pooled) {
  size_t gid = (size_t)blockIdx.x * blockDim.x + threadIdx.x;
  if (gid >= (size_t)NN * NH) return;
  int n = (int)(gid >> 6), c = (int)(gid & 63);
  atomicAdd(&pooled[batch[n] * NH + c], h[gid]);
}

__global__ void k_fc2(const float* __restrict__ z1, const float* __restrict__ W2,
                      const float* __restrict__ b2, float* __restrict__ z2) {
  int tid = threadIdx.x;
  if (tid >= BG * NC) return;
  int b = tid / NC, j = tid % NC;
  float s = b2[j];
  for (int k = 0; k < NH; ++k) s += z1[b * NH + k] * W2[k * NC + j];
  z2[b * NC + j] = s;
}

__global__ void k_bn_rows(const float* __restrict__ z, const float* __restrict__ g,
                          const float* __restrict__ beta, float* __restrict__ out,
                          int R, int C, int relu) {
  int c = threadIdx.x;
  if (c >= C) return;
  float a = 0.f, b = 0.f;
  for (int r = 0; r < R; ++r) { float v = z[r * C + c]; a += v; b += v * v; }
  float m = a / R, var = b / R - m * m, inv = rsqrtf(var + EPSV);
  for (int r = 0; r < R; ++r) {
    float v = (z[r * C + c] - m) * inv * g[c] + beta[c];
    if (relu) v = v > 0.f ? v : 0.f;
    out[r * C + c] = v;
  }
}

// ------------------------------------------------------------------ driver
static inline unsigned gb(size_t n) { return (unsigned)((n + 255) / 256); }

extern "C" void kernel_launch(void* const* d_in, const int* in_sizes, int n_in,
                              void* d_out, int out_size, void* d_ws, size_t ws_size,
                              hipStream_t stream) {
  // setup_inputs() dict order (sei/sea interleaved!)
  const float* x        = (const float*)d_in[0];
  const int*   ei       = (const int*)  d_in[1];
  const int*   batch    = (const int*)  d_in[2];
  const int*   sei[4]   = { (const int*)d_in[3], (const int*)d_in[5],
                            (const int*)d_in[7], (const int*)d_in[9] };
  const float* sea[4]   = { (const float*)d_in[4], (const float*)d_in[6],
                            (const float*)d_in[8], (const float*)d_in[10] };
  const float* W_in     = (const float*)d_in[11];
  const float* b_in     = (const float*)d_in[12];
  const float* W_gat    = (const float*)d_in[13];
  const float* b_gat    = (const float*)d_in[14];
  const float* att_src  = (const float*)d_in[15];
  const float* att_dst  = (const float*)d_in[16];
  const float* W_mlp    = (const float*)d_in[17];
  const float* b_mlp    = (const float*)d_in[18];
  const float* bn_gamma = (const float*)d_in[19];
  const float* bn_beta  = (const float*)d_in[20];
  const float* W_fc1    = (const float*)d_in[21];
  const float* b_fc1    = (const float*)d_in[22];
  const float* g_fc1    = (const float*)d_in[23];
  const float* beta_fc1 = (const float*)d_in[24];
  const float* W_fc2    = (const float*)d_in[25];
  const float* b_fc2    = (const float*)d_in[26];
  const float* g_fc2    = (const float*)d_in[27];
  const float* beta_fc2 = (const float*)d_in[28];
  float* out = (float*)d_out;

  // ---- workspace carving (floats; all chunk sizes keep 32B alignment)
  const size_t NF  = (size_t)NN * NH;   // 3.2M
  const size_t NFH = (size_t)NN * HF;   // 12.8M
  float* ws = (float*)d_ws;
  size_t o = 0;
  float* hbuf = ws + o; o += NF;
  float* acc  = ws + o; o += NF;
  float* xb0  = ws + o; o += NF;
  float* xb1  = ws + o; o += NF;
  float* xb2  = ws + o; o += NF;
  float* xb3  = ws + o; o += NF;
  float* t1   = ws + o; o += NF;
  float* t2   = ws + o; o += NF;
  float* t3   = ws + o; o += NF;
  float* hp   = ws + o; o += NFH;
  float* gz   = ws + o; o += NFH;
  float* mz   = ws + o; o += NF;
  float* als  = ws + o; o += (size_t)NN * HD;
  float* ald  = ws + o; o += (size_t)NN * HD;
  float* ebuf = ws + o; o += (size_t)ET * HD;
  float* mmax = ws + o; o += (size_t)NN * HD;
  float* den  = ws + o; o += (size_t)NN * HD;
  float* stats= ws + o; o += 2 * NH;
  float* pool = ws + o; o += (size_t)BG * NH;
  float* fc1z = ws + o; o += (size_t)BG * NH;
  float* fc2z = ws + o; o += (size_t)BG * NC;
  // packed f16 weights (fragment order)
  _Float16* wp_base = (_Float16*)(ws + o);
  size_t ho = 0;
  _Float16* wp_in  = wp_base + ho; ho += (size_t)NH * NH;           // 4096
  _Float16* wp_fc1 = wp_base + ho; ho += (size_t)NH * NH;           // 4096
  _Float16* wp_gat = wp_base + ho; ho += (size_t)12 * NH * HF;      // 12x16384
  _Float16* wp_mlp = wp_base + ho; ho += (size_t)12 * HF * NH;      // 12x16384
  float* xb[4] = { xb0, xb1, xb2, xb3 };
  float* tb[3] = { t1, t2, t3 };

  auto zero = [&](float* p, size_t n) {
    k_fill<<<gb(n), 256, 0, stream>>>(p, 0.0f, n);
  };
  auto pack = [&](const float* W, _Float16* Wp, int Kin, int Kout) {
    int total = (Kout >> 4) * (Kin >> 5) * 32;
    k_pack_w<<<gb((size_t)total), 256, 0, stream>>>(W, Wp, Kin, Kout);
  };

  // ---- pre-pack all weight matrices once (reused across all 3 blocks)
  pack(W_in,  wp_in,  NH, NH);
  pack(W_fc1, wp_fc1, NH, NH);
  for (int g = 0; g < 12; ++g) {
    pack(W_gat + (size_t)g * NH * HF, wp_gat + (size_t)g * NH * HF, NH, HF);
    pack(W_mlp + (size_t)g * HF * NH, wp_mlp + (size_t)g * HF * NH, HF, NH);
  }

  const int tilesM = NN >> 4;                       // 3125
  // ---- input projection: h = x @ W_in + b_in   (16x64 strip/wave)
  k_linear_t<NH, NH, 4><<<(tilesM + 7) / 8, 256, 0, stream>>>(
      x, wp_in, b_in, hbuf, NN);

  for (int i = 0; i < NSCB; ++i) {
    // scatter hop matrices
    zero(xb0, NF); k_scatter<<<gb((size_t)ESZ*NH), 256, 0, stream>>>(sei[0], sea[0], hbuf, xb0);
    for (int j = 1; j < 4; ++j) {
      zero(tb[j-1], NF);
      k_scatter<<<gb((size_t)ESZ*NH), 256, 0, stream>>>(sei[j], sea[j], hbuf, tb[j-1]);
      zero(xb[j], NF);
      k_scatter<<<gb((size_t)ESZ*NH), 256, 0, stream>>>(sei[0], sea[0], tb[j-1], xb[j]);
    }
    zero(acc, NF);
    for (int j = 0; j < 4; ++j) {
      int g = 4 * i + j;
      // hp = xj @ W_gat[g]  (64 -> 256): 4 waves per row strip
      k_linear_t<NH, HF, 4><<<(tilesM * 4 + 7) / 8, 256, 0, stream>>>(
          xb[j], wp_gat + (size_t)g * NH * HF, nullptr, hp, NN);
      // attention logits
      k_att_scores<<<gb((size_t)NN*HD), 256, 0, stream>>>(
          hp, att_src + (size_t)g * HF, att_dst + (size_t)g * HF, als, ald);
      // segment softmax over incoming edges
      k_fill<<<gb((size_t)NN*HD), 256, 0, stream>>>(mmax, -1e30f, (size_t)NN*HD);
      zero(den, (size_t)NN*HD);
      k_edge_pass1<<<gb((size_t)ET*HD), 256, 0, stream>>>(ei, als, ald, ebuf, mmax);
      k_edge_pass2<<<gb((size_t)ET*HD), 256, 0, stream>>>(ei, mmax, ebuf, den);
      // weighted aggregation
      zero(gz, NFH);
      k_edge_aggr<<<gb((size_t)ET*HF), 256, 0, stream>>>(ei, ebuf, den, hp, gz);
      // + b_gat, ELU
      k_bias_elu<<<gb(NFH), 256, 0, stream>>>(gz, b_gat + (size_t)g * HF);
      // mlp: 256 -> 64 (K fully unrolled: 32 wmma in the body)
      k_linear_t<HF, NH, 4><<<(tilesM + 7) / 8, 256, 0, stream>>>(
          gz, wp_mlp + (size_t)g * HF * NH, b_mlp + (size_t)g * NH, mz, NN);
      // BN (training stats) accumulated into acc
      k_bn_stats<<<NH, 256, 0, stream>>>(mz, stats, NN, NH);
      k_bn_apply_acc<<<gb(NF), 256, 0, stream>>>(
          mz, stats, bn_gamma + (size_t)g * NH, bn_beta + (size_t)g * NH, acc);
    }
    k_add<<<gb(NF), 256, 0, stream>>>(hbuf, acc, NF);
  }

  // ---- readout head
  zero(pool, (size_t)BG * NH);
  k_pool<<<gb(NF), 256, 0, stream>>>(batch, hbuf, pool);
  k_linear_t<NH, NH, 4><<<1, 256, 0, stream>>>(pool, wp_fc1, b_fc1, fc1z, BG);
  k_bn_rows<<<1, 64, 0, stream>>>(fc1z, g_fc1, beta_fc1, fc1z, BG, NH, 1);
  k_fc2<<<1, 640, 0, stream>>>(fc1z, W_fc2, b_fc2, fc2z);
  k_bn_rows<<<1, 32, 0, stream>>>(fc2z, g_fc2, beta_fc2, out, BG, NC, 0);
}
